// HaloAttn_36876589204203
// MI455X (gfx1250) — matrix operand order repaired and checked by reading, not compiled, not run on hardware
//
#include <hip/hip_runtime.h>
#include <hip/hip_bf16.h>

typedef __attribute__((ext_vector_type(16))) __bf16 v16bf;
typedef __attribute__((ext_vector_type(8)))  float  v8f;
typedef __attribute__((ext_vector_type(4)))  int    v4i;

union V16U {
    unsigned short u[16];
    unsigned int   w[8];
    uint4          q4[2];
    v16bf          v;
};

// round-half-up bf16: 2 VALU per scalar, 3 per packed pair (v_perm_b32)
__device__ __forceinline__ unsigned short f2bf(float f) {
    return (unsigned short)((__float_as_uint(f) + 0x8000u) >> 16);
}
__device__ __forceinline__ unsigned int pack2bf(float lo, float hi) {
    return __builtin_amdgcn_perm(__float_as_uint(hi) + 0x8000u,
                                 __float_as_uint(lo) + 0x8000u, 0x07060302u);
}
__device__ __forceinline__ float bf2f(unsigned short s) {
    return __uint_as_float(((unsigned int)s) << 16);
}

// ---- gfx1250 async global->LDS copy path (ASYNCcnt) ----
#if defined(__has_builtin)
#if __has_builtin(__builtin_amdgcn_global_load_async_to_lds_b128) && \
    __has_builtin(__builtin_amdgcn_s_wait_asynccnt)
#define HAVE_ASYNC_LDS 1
#endif
#endif
#ifndef HAVE_ASYNC_LDS
#define HAVE_ASYNC_LDS 0
#endif

typedef __attribute__((address_space(1))) v4i* g_v4i_ptr;
typedef __attribute__((address_space(3))) v4i* l_v4i_ptr;

__device__ __forceinline__ void copy_b128_g2l(const unsigned short* g, unsigned short* l) {
#if HAVE_ASYNC_LDS
    __builtin_amdgcn_global_load_async_to_lds_b128(
        (g_v4i_ptr)(unsigned long long)(uintptr_t)g,
        (l_v4i_ptr)(unsigned int)(uintptr_t)l, 0, 0);
#else
    *(uint4*)l = *(const uint4*)g;
#endif
}
__device__ __forceinline__ void async_fence() {
#if HAVE_ASYNC_LDS
    __builtin_amdgcn_s_wait_asynccnt(0);
#endif
}

#define HW    4096      /* 64*64 */
#define SCALE 0.25f     /* DH^-0.5 */

// ---------------------------------------------------------------------------
// Kernel 0a: x f32 -> bf16 (same [b][c][hw] layout), 4 elems/thread.
// ---------------------------------------------------------------------------
__global__ __launch_bounds__(256) void cvt_x_kernel(const float* __restrict__ x,
                                                    unsigned short* __restrict__ xbf) {
    const int i = blockIdx.x * 256 + threadIdx.x;       // 2,097,152 tasks
    const float4 f = ((const float4*)x)[i];
    uint2 o;
    o.x = pack2bf(f.x, f.y);
    o.y = pack2bf(f.z, f.w);
    ((uint2*)xbf)[i] = o;
}

// ---------------------------------------------------------------------------
// Kernel 0b: pre-pack weights into WMMA A-fragment order:
// wpack[frag = (mt*8 + kstep)*32 + lane][16 bf16], element i -> k = k0+g*16+half*8+j.
// ---------------------------------------------------------------------------
__global__ __launch_bounds__(256) void pack_w_kernel(const float* __restrict__ q_w,
                                                     const float* __restrict__ kv_w,
                                                     unsigned short* __restrict__ wpack) {
    const int t = blockIdx.x * 256 + threadIdx.x;       // 8192 fragments
    const int lane = t & 31, ks = (t >> 5) & 7, mt = t >> 8;
    const int lanelo = lane & 15, half = lane >> 4;
    const int m = mt * 16 + lanelo;
    const float* wr = (m < 128) ? q_w + (size_t)m * 256
                                : kv_w + (size_t)(m - 128) * 256;
    const int k0 = ks * 32;
    const float4 f0 = *(const float4*)(wr + k0 + half * 8);
    const float4 f1 = *(const float4*)(wr + k0 + half * 8 + 4);
    const float4 f2 = *(const float4*)(wr + k0 + 16 + half * 8);
    const float4 f3 = *(const float4*)(wr + k0 + 16 + half * 8 + 4);
    uint4 o0, o1;
    o0.x = pack2bf(f0.x, f0.y); o0.y = pack2bf(f0.z, f0.w);
    o0.z = pack2bf(f1.x, f1.y); o0.w = pack2bf(f1.z, f1.w);
    o1.x = pack2bf(f2.x, f2.y); o1.y = pack2bf(f2.z, f2.w);
    o1.z = pack2bf(f3.x, f3.y); o1.w = pack2bf(f3.z, f3.w);
    uint4* dst = (uint4*)(wpack + (size_t)t * 16);
    dst[0] = o0;
    dst[1] = o1;
}

// ---------------------------------------------------------------------------
// Kernel 1: fused QKV projection, pixel-major output qkv[pixel][512].
// GEMM M=512 x N=32768 x K=256 (bf16 WMMA, f32 acc). Block = 512 thr
// (16 waves) owns one 16-pixel N-tile; bf16 x tile staged/transposed in LDS
// once; each wave computes M-tiles (w) and (w+16). Inner loop: pure b128
// loads + WMMA, no conversions.
// ---------------------------------------------------------------------------
__global__ __launch_bounds__(512) void qkv_gemm_kernel(
    const unsigned short* __restrict__ xbf, const unsigned short* __restrict__ wpack,
    unsigned short* __restrict__ qkv) {
    __shared__ __align__(16) unsigned short xt[16][256];   // [pixel][k] bf16

    const int tid = threadIdx.x;
    const int lane = tid & 31, lanelo = lane & 15, half = lane >> 4;
    const int w = tid >> 5;                  // wave 0..15
    const int p0 = blockIdx.x << 4;          // pixel base
    const int b = p0 >> 12, sp = p0 & 4095;  // tile never crosses an image

    // stage x tile: one b128 load + 8-way bf16 scatter-transpose per thread
    {
        const int c = tid >> 1, pxh = tid & 1;
        const uint4 v = *(const uint4*)&xbf[(size_t)b * 256 * HW + (size_t)c * HW + sp + pxh * 8];
        const unsigned int wv[4] = {v.x, v.y, v.z, v.w};
#pragma unroll
        for (int j = 0; j < 4; ++j) {
            xt[pxh * 8 + 2 * j][c]     = (unsigned short)(wv[j] & 0xFFFFu);
            xt[pxh * 8 + 2 * j + 1][c] = (unsigned short)(wv[j] >> 16);
        }
    }
    const unsigned short* wp0 = wpack + ((size_t)(w * 8) * 32 + lane) * 16;
    const unsigned short* wp1 = wpack + ((size_t)((w + 16) * 8) * 32 + lane) * 16;
    __builtin_prefetch(wp0, 0, 3);           // global_prefetch_b8 path
    __builtin_prefetch(wp1, 0, 3);
    __syncthreads();

    v8f acc0 = {}, acc1 = {};
#pragma unroll
    for (int ks = 0; ks < 8; ++ks) {
        V16U a0, a1, bm;
        bm.q4[0] = *(const uint4*)&xt[lanelo][ks * 32 + half * 8];
        bm.q4[1] = *(const uint4*)&xt[lanelo][ks * 32 + 16 + half * 8];
        a0.q4[0] = *(const uint4*)(wp0 + ks * 512);
        a0.q4[1] = *(const uint4*)(wp0 + ks * 512 + 8);
        a1.q4[0] = *(const uint4*)(wp1 + ks * 512);
        a1.q4[1] = *(const uint4*)(wp1 + ks * 512 + 8);
        acc0 = __builtin_amdgcn_wmma_f32_16x16x32_bf16(false, a0.v, false, bm.v,
                                                       (short)0, acc0, false, false);
        acc1 = __builtin_amdgcn_wmma_f32_16x16x32_bf16(false, a1.v, false, bm.v,
                                                       (short)0, acc1, false, false);
    }
    // D: lane holds pixel p0+lanelo, 8 consecutive channels -> one b128 each
    unsigned short* op = qkv + (size_t)(p0 + lanelo) * 512;
    uint4 s0, s1;
    s0.x = pack2bf(acc0[0], acc0[1]); s0.y = pack2bf(acc0[2], acc0[3]);
    s0.z = pack2bf(acc0[4], acc0[5]); s0.w = pack2bf(acc0[6], acc0[7]);
    s1.x = pack2bf(acc1[0], acc1[1]); s1.y = pack2bf(acc1[2], acc1[3]);
    s1.z = pack2bf(acc1[4], acc1[5]); s1.w = pack2bf(acc1[6], acc1[7]);
    *(uint4*)&op[(w << 4) + (half << 3)]        = s0;   // tile w
    *(uint4*)&op[((w + 16) << 4) + (half << 3)] = s1;   // tile w+16
}

// ---------------------------------------------------------------------------
// Kernel 2: halo attention per (b, head, block). 4 waves; each wave owns a
// 16-query strip. S kept in registers (13 v8f), softmax via shfl_xor.
// qkv layout: [b*4096 + y*64 + x][512]; ch 0..127 q (h*16+d),
// 128..511 kv (128 + h*48 + {d | 16+dv}).
// ---------------------------------------------------------------------------
#define SMEM_BYTES 47104
#define OFF_Q   0        /* 64x32 bf16  = 4096 B  */
#define OFF_V   4096     /* vT 32x224   = 14336 B */
#define OFF_POS 18432    /* 64x28 f32   = 7168 B  (overwritten by P) */
#define OFF_K   25600    /* 208x32 bf16 = 13312 B (overwritten by P) */
#define OFF_P   18432    /* 64x224 bf16 = 28672 B -> ends at 47104   */

__global__ __launch_bounds__(128) void halo_attn_kernel(
    const unsigned short* __restrict__ qkv,
    const float* __restrict__ height_rel, const float* __restrict__ width_rel,
    float* __restrict__ out) {
    __shared__ __align__(16) char smem[SMEM_BYTES];
    unsigned short* qs  = (unsigned short*)(smem + OFF_Q);   // [64][32]
    unsigned short* vt  = (unsigned short*)(smem + OFF_V);   // [32][224]
    float*          pos = (float*)(smem + OFF_POS);          // [64][28]
    unsigned short* ks  = (unsigned short*)(smem + OFF_K);   // [208][32]
    unsigned short* ps  = (unsigned short*)(smem + OFF_P);   // [64][224]

    const int tid = threadIdx.x;
    const int lane = tid & 31, lanelo = lane & 15, half = lane >> 4;
    const int strip = tid >> 5;
    const int nblk = blockIdx.x & 63, zb = blockIdx.x >> 6;
    const int h = zb & 7, b = zb >> 3;
    const int nhb = nblk >> 3, nwb = nblk & 7;
    const size_t pbase = (size_t)b * HW;
    const uint4 zero4 = {0u, 0u, 0u, 0u};

    for (int i = tid; i < SMEM_BYTES / 16; i += 128) ((uint4*)smem)[i] = zero4;
    __syncthreads();

    // q tile: 64 queries x 16 ch, contiguous b128 copies (async path)
    for (int i = tid; i < 64 * 2; i += 128) {
        const int qi = i >> 1, part = i & 1;
        const int y = nhb * 8 + (qi >> 3), xx = nwb * 8 + (qi & 7);
        copy_b128_g2l(&qkv[(pbase + y * 64 + xx) * 512 + h * 16 + part * 8],
                      &qs[qi * 32 + part * 8]);
    }
    // k window: 196 keys x 16 ch; OOB lanes skip the copy, LDS pre-zeroed
    for (int i = tid; i < 196 * 2; i += 128) {
        const int key = i >> 1, part = i & 1;
        const int y = nhb * 8 + key / 14 - 3, xx = nwb * 8 + key % 14 - 3;
        if (y >= 0 && y < 64 && xx >= 0 && xx < 64)
            copy_b128_g2l(&qkv[(pbase + y * 64 + xx) * 512 + 128 + h * 48 + part * 8],
                          &ks[key * 32 + part * 8]);
    }
    // v window: load 8 ch at a time, scatter-transpose into vt[dv][key]
    for (int i = tid; i < 196 * 4; i += 128) {
        const int key = i >> 2, g = i & 3;
        const int y = nhb * 8 + key / 14 - 3, xx = nwb * 8 + key % 14 - 3;
        uint4 val = zero4;
        if (y >= 0 && y < 64 && xx >= 0 && xx < 64)
            val = *(const uint4*)&qkv[(pbase + y * 64 + xx) * 512 + 144 + h * 48 + g * 8];
        const unsigned int wv[4] = {val.x, val.y, val.z, val.w};
#pragma unroll
        for (int j = 0; j < 4; ++j) {
            vt[(g * 8 + 2 * j) * 224 + key]     = (unsigned short)(wv[j] & 0xFFFFu);
            vt[(g * 8 + 2 * j + 1) * 224 + key] = (unsigned short)(wv[j] >> 16);
        }
    }
    async_fence();          // s_wait_asynccnt 0 (if async path in use)
    __syncthreads();

    // decomposed relative-position tables: hlog[qi][ky], wlog[qi][kx]
    for (int i = tid; i < 64 * 28; i += 128) {
        const int qi = i / 28, j = i % 28;
        const float* rrow = (j < 14) ? (height_rel + (13 - (qi >> 3) + j) * 16)
                                     : (width_rel  + (13 - (qi & 7) + (j - 14)) * 16);
        V16U qr;
        qr.q4[0] = *(const uint4*)&qs[qi * 32];
        qr.q4[1] = *(const uint4*)&qs[qi * 32 + 8];
        float s = 0.f;
#pragma unroll
        for (int d = 0; d < 8; ++d) {
            s += bf2f((unsigned short)(qr.w[d] & 0xFFFFu)) * rrow[2 * d];
            s += bf2f((unsigned short)(qr.w[d] >> 16)) * rrow[2 * d + 1];
        }
        pos[qi * 28 + j] = s;
    }
    __syncthreads();

    // ---- S = q k^T  (strip 16 x 208, K padded to 32) ----
    V16U aq;
    aq.q4[0] = *(const uint4*)&qs[(strip * 16 + lanelo) * 32 + half * 8];
    aq.q4[1] = *(const uint4*)&qs[(strip * 16 + lanelo) * 32 + 16 + half * 8];
    v8f S[13];
#pragma unroll
    for (int t = 0; t < 13; ++t) {
        V16U bk;
        bk.q4[0] = *(const uint4*)&ks[(t * 16 + lanelo) * 32 + half * 8];
        bk.q4[1] = *(const uint4*)&ks[(t * 16 + lanelo) * 32 + 16 + half * 8];
        v8f z = {};
        S[t] = __builtin_amdgcn_wmma_f32_16x16x32_bf16(false, aq.v, false, bk.v,
                                                       (short)0, z, false, false);
    }
    // scale + relative position bias, mask padded keys
#pragma unroll
    for (int t = 0; t < 13; ++t) {
        const int n = t * 16 + lanelo;
        const bool valid = n < 196;
        const int ky = n / 14, kx = n % 14;
#pragma unroll
        for (int r = 0; r < 8; ++r) {
            const int qi = strip * 16 + r + half * 8;
            S[t][r] = valid ? (S[t][r] * SCALE + pos[qi * 28 + ky] + pos[qi * 28 + 14 + kx])
                            : -3.0e38f;
        }
    }
    // ---- softmax over 196 keys (16 lanes x 13 tiles per row) ----
    float inv[8];
#pragma unroll
    for (int r = 0; r < 8; ++r) {
        float m = -3.0e38f;
#pragma unroll
        for (int t = 0; t < 13; ++t) m = fmaxf(m, S[t][r]);
#pragma unroll
        for (int off = 1; off < 16; off <<= 1) m = fmaxf(m, __shfl_xor(m, off, 32));
        float s = 0.f;
#pragma unroll
        for (int t = 0; t < 13; ++t) { const float e = __expf(S[t][r] - m); S[t][r] = e; s += e; }
#pragma unroll
        for (int off = 1; off < 16; off <<= 1) s += __shfl_xor(s, off, 32);
        inv[r] = 1.0f / s;
    }
    __syncthreads();   // all waves done reading pos/k before P overwrites them
#pragma unroll
    for (int t = 0; t < 13; ++t)
#pragma unroll
        for (int r = 0; r < 8; ++r)
            ps[(strip * 16 + r + half * 8) * 224 + t * 16 + lanelo] = f2bf(S[t][r] * inv[r]);
    if (lane < 16) {                     // zero K-pad columns 208..223 of this strip
        *(uint4*)&ps[(strip * 16 + lane) * 224 + 208] = zero4;
        *(uint4*)&ps[(strip * 16 + lane) * 224 + 216] = zero4;
    }
    __syncthreads();

    // ---- O = P v  (strip 16 x 32, K = 224 in 7 steps) ----
    v8f O0 = {}, O1 = {};
#pragma unroll
    for (int t2 = 0; t2 < 7; ++t2) {
        V16U ap, bv0, bv1;
        const int abase = (strip * 16 + lanelo) * 224 + t2 * 32;
        ap.q4[0] = *(const uint4*)&ps[abase + half * 16];
        ap.q4[1] = *(const uint4*)&ps[abase + half * 16 + 8];
        const int vb0 = lanelo * 224 + t2 * 32 + half * 16;
        const int vb1 = (16 + lanelo) * 224 + t2 * 32 + half * 16;
        bv0.q4[0] = *(const uint4*)&vt[vb0]; bv0.q4[1] = *(const uint4*)&vt[vb0 + 8];
        bv1.q4[0] = *(const uint4*)&vt[vb1]; bv1.q4[1] = *(const uint4*)&vt[vb1 + 8];
        O0 = __builtin_amdgcn_wmma_f32_16x16x32_bf16(false, ap.v, false, bv0.v,
                                                     (short)0, O0, false, false);
        O1 = __builtin_amdgcn_wmma_f32_16x16x32_bf16(false, ap.v, false, bv1.v,
                                                     (short)0, O1, false, false);
    }
    // scatter to out[b, h*32+dv, y, x]
#pragma unroll
    for (int r = 0; r < 8; ++r) {
        const int qi = strip * 16 + r + half * 8;
        const int y = nhb * 8 + (qi >> 3), xx = nwb * 8 + (qi & 7);
        const size_t ob = (size_t)b * 256 * HW + (size_t)y * 64 + xx;
        out[ob + (size_t)(h * 32 + lanelo) * HW]      = O0[r];
        out[ob + (size_t)(h * 32 + 16 + lanelo) * HW] = O1[r];
    }
}

extern "C" void kernel_launch(void* const* d_in, const int* in_sizes, int n_in,
                              void* d_out, int out_size, void* d_ws, size_t ws_size,
                              hipStream_t stream) {
    const float* x          = (const float*)d_in[0];
    const float* q_w        = (const float*)d_in[1];
    const float* kv_w       = (const float*)d_in[2];
    const float* height_rel = (const float*)d_in[3];
    const float* width_rel  = (const float*)d_in[4];
    float* out = (float*)d_out;
    // workspace layout
    unsigned short* qkv   = (unsigned short*)d_ws;                       // 33,554,432 B
    unsigned short* xbf   = (unsigned short*)((char*)d_ws + 33554432);   // 16,777,216 B
    unsigned short* wpack = (unsigned short*)((char*)d_ws + 50331648);   //    262,144 B

    cvt_x_kernel<<<8192, 256, 0, stream>>>(x, xbf);
    pack_w_kernel<<<32, 256, 0, stream>>>(q_w, kv_w, wpack);
    qkv_gemm_kernel<<<2048, 512, 0, stream>>>(xbf, wpack, qkv);
    halo_attn_kernel<<<4096, 128, 0, stream>>>(qkv, height_rel, width_rel, out);
}
